// WmDeformerBlock_69569880261286
// MI455X (gfx1250) — matrix operand order repaired and checked by reading, not compiled, not run on hardware
//
#include <hip/hip_runtime.h>
#include <hip/hip_fp16.h>
#include <stdint.h>

typedef __attribute__((ext_vector_type(16))) _Float16 v16h;
typedef __attribute__((ext_vector_type(8)))  _Float16 v8h;
typedef __attribute__((ext_vector_type(8)))  float    v8f;
typedef __attribute__((ext_vector_type(4)))  int      v4i;

#define GAT_HEADS 6
#define NEG_SLOPE 0.2f
#define SELU_SCALE 1.0507009873554805f
#define SELU_ALPHA 1.6732632423543772f

#if defined(__has_builtin)
#if __has_builtin(__builtin_amdgcn_global_load_async_to_lds_b128) && \
    __has_builtin(__builtin_amdgcn_s_wait_asynccnt)
#define USE_ASYNC_LDS 1
#endif
#endif

#ifdef USE_ASYNC_LDS
typedef __attribute__((address_space(1))) v4i* gas_v4i;   // global (prints as "__device__")
typedef __attribute__((address_space(3))) v4i* las_v4i;   // LDS
#endif

// ---------------------------------------------------------------------------
// WMMA GEMM on pre-padded f16 operands:
//   Hout[n][m] = sum_k X[n*ldx+k] * W[m*ldw+k]
// X: (>=ceil64(N)) x ldx f16, zero-padded; W: (>=ceil64(M)) x ldw f16, padded.
// Block = 128 threads (4 waves). Block tile 64 rows x 64 cols, K-step 64.
// Each wave owns a 16-row strip and 4 accumulators -> 8 WMMAs per barrier,
// A fragment reused 4x. Tiles staged with global_load_async_to_lds_b128 into
// DOUBLE-BUFFERED LDS: copies for tile i+1 are issued before computing tile i,
// then s_wait_asynccnt 8 retires exactly tile i's 8 per-wave async loads
// (async loads complete in order), overlapping DMA with WMMA work.
// Fragment layouts used here (wave32, contiguous K, same order for A and B):
//   A: row = lane&15; v0..7: K = 16*(lane>>4)+{0..15}
//   B: col = lane&15; v0..7: K = 16*(lane>>4)+{0..15}
//   D: col = lane&15; vj: row = j + 8*(lane>>4)
// ---------------------------------------------------------------------------
__global__ __launch_bounds__(128)
void gemm_f16_wmma(const _Float16* __restrict__ X, int ldx,
                   const _Float16* __restrict__ W, int ldw,
                   float* __restrict__ Hout, int N, int M, int Kp)
{
    __shared__ _Float16 Xs[2][64][72];   // 144B row stride (16B aligned)
    __shared__ _Float16 Ws[2][64][72];

    const int tid  = threadIdx.x;
    const int wave = tid >> 5;
    const int lane = tid & 31;
    const int half = lane >> 4;
    const int r    = lane & 15;
    const int n0   = blockIdx.x * 64;
    const int mblk = blockIdx.y * 64;

    // stage two 64x64 f16 tiles (X and W) in 16B chunks: chunk c -> row c>>3,
    // col 8*(c&7). 4 chunk-iterations x 2 copies = 8 async issues per wave.
    auto stage = [&](int buf, int k0) {
        for (int c = tid; c < 512; c += 128) {
            int row = c >> 3, ck = (c & 7) * 8;
            const _Float16* gx = &X[(size_t)(n0 + row) * ldx + k0 + ck];
            const _Float16* gw = &W[(size_t)(mblk + row) * ldw + k0 + ck];
            _Float16* lx = &Xs[buf][row][ck];
            _Float16* lw = &Ws[buf][row][ck];
#ifdef USE_ASYNC_LDS
            __builtin_amdgcn_global_load_async_to_lds_b128((gas_v4i)gx, (las_v4i)lx, 0, 0);
            __builtin_amdgcn_global_load_async_to_lds_b128((gas_v4i)gw, (las_v4i)lw, 0, 0);
#else
            *(v8h*)lx = *(const v8h*)gx;
            *(v8h*)lw = *(const v8h*)gw;
#endif
        }
    };

    v8f acc[4] = {};
    const int nk = Kp >> 6;
    stage(0, 0);                                     // prologue: tile 0 -> buf 0

    for (int i = 0; i < nk; ++i) {
        const int buf = i & 1;
        const bool has_next = (i + 1) < nk;
        if (has_next) {
            __builtin_prefetch(&W[(size_t)mblk * ldw + (i + 1) * 64], 0, 1);
            stage(buf ^ 1, (i + 1) * 64);            // DMA tile i+1 under tile i's math
        }
#ifdef USE_ASYNC_LDS
        if (has_next) __builtin_amdgcn_s_wait_asynccnt(8);   // tile i landed
        else          __builtin_amdgcn_s_wait_asynccnt(0);
#endif
        __syncthreads();

        const _Float16* xrow = &Xs[buf][wave * 16 + r][0];
        #pragma unroll
        for (int kh = 0; kh < 2; ++kh) {
            v16h a;
            #pragma unroll
            for (int j = 0; j < 8; ++j) {
                a[j]     = xrow[kh * 32 + half * 16 + j];
                a[8 + j] = xrow[kh * 32 + half * 16 + 8 + j];
            }
            #pragma unroll
            for (int ct = 0; ct < 4; ++ct) {
                const _Float16* wrow = &Ws[buf][ct * 16 + r][0];
                v16h b;
                #pragma unroll
                for (int j = 0; j < 16; ++j)
                    b[j] = wrow[kh * 32 + half * 16 + j];
                acc[ct] = __builtin_amdgcn_wmma_f32_16x16x32_f16(
                              false, a, false, b, (short)0, acc[ct], false, false);
            }
        }
        __syncthreads();   // done reading buf before it is overwritten (i+2)
    }

    #pragma unroll
    for (int ct = 0; ct < 4; ++ct) {
        int gm = mblk + ct * 16 + r;
        if (gm < M) {
            #pragma unroll
            for (int j = 0; j < 8; ++j) {
                int gn = n0 + wave * 16 + half * 8 + j;
                if (gn < N) Hout[(size_t)gn * M + gm] = acc[ct][j];
            }
        }
    }
}

// ---------------------------------------------------------------------------
// Conversion / padding kernels (f32 -> padded f16)
// ---------------------------------------------------------------------------
__global__ void cvt_pad_f16(const float* __restrict__ src, int R, int K,
                            _Float16* __restrict__ dst, int Kp, int Rpad)
{
    int idx = blockIdx.x * blockDim.x + threadIdx.x;
    if (idx >= Rpad * Kp) return;
    int rr = idx / Kp, k = idx % Kp;
    float v = (rr < R && k < K) ? src[(size_t)rr * K + k] : 0.0f;
    dst[idx] = (_Float16)v;
}

__global__ void fill_f16(_Float16* p, int n)
{ int i = blockIdx.x * blockDim.x + threadIdx.x; if (i < n) p[i] = (_Float16)0.0f; }

__global__ void copy_cols_f16(_Float16* __restrict__ dst, int dstLd, int dstOff,
                              const float* __restrict__ src, int srcLd, int srcOff,
                              int ncols, int N)
{
    int idx = blockIdx.x * blockDim.x + threadIdx.x;
    if (idx >= N * ncols) return;
    int n = idx / ncols, c = idx % ncols;
    dst[(size_t)n * dstLd + dstOff + c] = (_Float16)src[(size_t)n * srcLd + srcOff + c];
}

// ---------------------------------------------------------------------------
// Elementwise / GAT helper kernels
// ---------------------------------------------------------------------------
__global__ void bias_selu_kernel(const float* __restrict__ H, const float* __restrict__ b,
                                 float* __restrict__ out, int N, int C)
{
    int idx = blockIdx.x * blockDim.x + threadIdx.x;
    if (idx >= N * C) return;
    int c = idx % C;
    float v = H[idx] + b[c];
    out[idx] = (v > 0.0f) ? SELU_SCALE * v
                          : SELU_SCALE * SELU_ALPHA * (__expf(v) - 1.0f);
}

__global__ void alpha_kernel(const float* __restrict__ h,
                             const float* __restrict__ a_src, const float* __restrict__ a_dst,
                             float* __restrict__ asrc, float* __restrict__ adst, int N, int C)
{
    int idx = blockIdx.x * blockDim.x + threadIdx.x;
    if (idx >= N * GAT_HEADS) return;
    int n = idx / GAT_HEADS, hh = idx % GAT_HEADS;
    const float* hr = h + (size_t)n * GAT_HEADS * C + (size_t)hh * C;
    const float* as = a_src + (size_t)hh * C;
    const float* ad = a_dst + (size_t)hh * C;
    float s0 = 0.0f, s1 = 0.0f;
    for (int c = 0; c < C; ++c) { float v = hr[c]; s0 += v * as[c]; s1 += v * ad[c]; }
    asrc[idx] = s0; adst[idx] = s1;
}

__global__ void fill_f32(float* p, float v, int n)
{ int i = blockIdx.x * blockDim.x + threadIdx.x; if (i < n) p[i] = v; }

__global__ void fill_u32(unsigned* p, unsigned v, int n)
{ int i = blockIdx.x * blockDim.x + threadIdx.x; if (i < n) p[i] = v; }

__device__ __forceinline__ unsigned f2ord(float f)
{ unsigned u = __float_as_uint(f); return (u & 0x80000000u) ? ~u : (u | 0x80000000u); }
__device__ __forceinline__ float ord2f(unsigned u)
{ return (u & 0x80000000u) ? __uint_as_float(u & 0x7FFFFFFFu) : __uint_as_float(~u); }

__device__ __forceinline__ void edge_ends(const int* eidx, int E, long long e, int& s, int& d)
{
    if (e < E) { s = eidx[e]; d = eidx[E + e]; }
    else       { s = d = (int)(e - E); }         // self loops appended
}

__global__ void edge_max_kernel(const int* __restrict__ eidx, int E, int N,
                                const float* __restrict__ asrc, const float* __restrict__ adst,
                                unsigned* __restrict__ emax)
{
    int idx = blockIdx.x * blockDim.x + threadIdx.x;
    int ET = E + N;
    if (idx >= ET * GAT_HEADS) return;
    long long e = idx / GAT_HEADS; int hh = idx % GAT_HEADS;
    int s, d; edge_ends(eidx, E, e, s, d);
    float ev = asrc[s * GAT_HEADS + hh] + adst[d * GAT_HEADS + hh];
    ev = (ev > 0.0f) ? ev : NEG_SLOPE * ev;
    atomicMax(&emax[d * GAT_HEADS + hh], f2ord(ev));
}

__global__ void edge_den_kernel(const int* __restrict__ eidx, int E, int N,
                                const float* __restrict__ asrc, const float* __restrict__ adst,
                                const unsigned* __restrict__ emax, float* __restrict__ denom)
{
    int idx = blockIdx.x * blockDim.x + threadIdx.x;
    int ET = E + N;
    if (idx >= ET * GAT_HEADS) return;
    long long e = idx / GAT_HEADS; int hh = idx % GAT_HEADS;
    int s, d; edge_ends(eidx, E, e, s, d);
    int nh = d * GAT_HEADS + hh;
    float ev = asrc[s * GAT_HEADS + hh] + adst[nh];
    ev = (ev > 0.0f) ? ev : NEG_SLOPE * ev;
    atomicAdd(&denom[nh], __expf(ev - ord2f(emax[nh])));
}

__global__ void edge_msg_kernel(const int* __restrict__ eidx, int E, int N,
                                const float* __restrict__ asrc, const float* __restrict__ adst,
                                const unsigned* __restrict__ emax, const float* __restrict__ denom,
                                const float* __restrict__ h, float* __restrict__ acc, int C)
{
    int cv = (C + 3) >> 2;
    long long idx = (long long)blockIdx.x * blockDim.x + threadIdx.x;
    long long tot = (long long)(E + N) * GAT_HEADS * cv;
    if (idx >= tot) return;
    int cb = (int)(idx % cv); long long t = idx / cv;
    int hh = (int)(t % GAT_HEADS); long long e = t / GAT_HEADS;
    int s, d; edge_ends(eidx, E, e, s, d);
    int nh = d * GAT_HEADS + hh;
    float ev = asrc[s * GAT_HEADS + hh] + adst[nh];
    ev = (ev > 0.0f) ? ev : NEG_SLOPE * ev;
    float alpha = __expf(ev - ord2f(emax[nh])) / (denom[nh] + 1e-16f);
    const float* hp = h   + (size_t)s * GAT_HEADS * C + (size_t)hh * C;
    float*       ap = acc + (size_t)d * GAT_HEADS * C + (size_t)hh * C;
    int c0 = cb * 4;
    #pragma unroll
    for (int j = 0; j < 4; ++j) {
        int c = c0 + j;
        if (c < C) atomicAdd(&ap[c], hp[c] * alpha);
    }
}

__global__ void finalize_feat(const float* __restrict__ acc, const float* __restrict__ b,
                              float* __restrict__ out, int N, int C)
{
    int idx = blockIdx.x * blockDim.x + threadIdx.x;
    if (idx >= N * C) return;
    int n = idx / C, c = idx % C;
    float s = 0.0f;
    #pragma unroll
    for (int hh = 0; hh < GAT_HEADS; ++hh)
        s += acc[(size_t)n * GAT_HEADS * C + (size_t)hh * C + c];
    float v = s * (1.0f / GAT_HEADS) + b[c];
    out[idx] = (v > 0.0f) ? SELU_SCALE * v
                          : SELU_SCALE * SELU_ALPHA * (__expf(v) - 1.0f);
}

__global__ void finalize_coord(const float* __restrict__ acc, const float* __restrict__ b,
                               const float* __restrict__ data, float* __restrict__ out, int N)
{
    int n = blockIdx.x * blockDim.x + threadIdx.x;
    if (n >= N) return;
    float c0 = 0.0f, c1 = 0.0f;
    #pragma unroll
    for (int hh = 0; hh < GAT_HEADS; ++hh) {
        c0 += acc[n * 12 + hh * 2 + 0];
        c1 += acc[n * 12 + hh * 2 + 1];
    }
    c0 = c0 * (1.0f / GAT_HEADS) + b[0];
    c1 = c1 * (1.0f / GAT_HEADS) + b[1];
    float d0 = data[n * 10 + 0], d1 = data[n * 10 + 1];
    if (d0 == 1.0f) c0 = 1.0f; else if (d0 == 0.0f) c0 = 0.0f;   // up / down
    if (d1 == 0.0f) c1 = 0.0f; else if (d1 == 1.0f) c1 = 1.0f;   // left / right
    out[n * 2 + 0] = c0;
    out[n * 2 + 1] = c1;
}

// ---------------------------------------------------------------------------
// Host orchestration
// ---------------------------------------------------------------------------
extern "C" void kernel_launch(void* const* d_in, const int* in_sizes, int n_in,
                              void* d_out, int out_size, void* d_ws, size_t ws_size,
                              hipStream_t stream)
{
    const float* data  = (const float*)d_in[0];
    const int*   eidx  = (const int*)  d_in[1];
    const float* lin_w = (const float*)d_in[2];
    const float* lin_b = (const float*)d_in[3];

    struct Gat { const float *W, *as, *ad, *b; int fin, fout; };
    const int fins[8]  = {258, 258, 516, 516, 262, 262, 136, 136};
    const int fouts[8] = {2, 512, 2, 256, 2, 128, 2, 32};
    Gat g[8];
    for (int i = 0; i < 8; ++i) {
        g[i].W  = (const float*)d_in[4 + 4 * i];
        g[i].as = (const float*)d_in[5 + 4 * i];
        g[i].ad = (const float*)d_in[6 + 4 * i];
        g[i].b  = (const float*)d_in[7 + 4 * i];
        g[i].fin = fins[i]; g[i].fout = fouts[i];
    }

    const int N = in_sizes[0] / 10;     // 10000
    const int E = in_sizes[1] / 2;      // 80000
    auto r64  = [](int v) { return (v + 63) & ~63; };
    const int Npad = r64(N);            // 10048

    // carve workspace
    char* wp = (char*)d_ws;
    auto carve = [&](size_t bytes) -> void* {
        void* p = (void*)wp; wp += (bytes + 255) & ~(size_t)255; return p;
    };
    float*     h_buf    = (float*)    carve((size_t)N * 3072 * 4);
    float*     acc_buf  = (float*)    carve((size_t)N * 3072 * 4);
    float*     feat_buf = (float*)    carve((size_t)N * 512 * 4);
    _Float16*  xh_buf   = (_Float16*) carve((size_t)Npad * 576 * 2);  // padded f16 X
    _Float16*  wh_buf   = (_Float16*) carve((size_t)3072 * 320 * 2);  // padded f16 W (max)
    float*     coord1   = (float*)    carve((size_t)N * 2 * 4);
    float*     coord2   = (float*)    carve((size_t)N * 2 * 4);
    float*     coord3   = (float*)    carve((size_t)N * 2 * 4);
    float*     asrc     = (float*)    carve((size_t)N * GAT_HEADS * 4);
    float*     adst     = (float*)    carve((size_t)N * GAT_HEADS * 4);
    unsigned*  emax     = (unsigned*) carve((size_t)N * GAT_HEADS * 4);
    float*     denom    = (float*)    carve((size_t)N * GAT_HEADS * 4);

    auto cdiv = [](long long a, long long b) { return (int)((a + b - 1) / b); };

    // xh_buf holds the current layer's concatenated input, padded f16 (ld = Kp)
    auto begin_x = [&](int Kp) {
        long long tot = (long long)Npad * Kp;
        fill_f16<<<cdiv(tot, 256), 256, 0, stream>>>(xh_buf, (int)tot);
    };
    auto put_x = [&](int Kp, int off, const float* src, int srcLd, int srcOff, int nc) {
        copy_cols_f16<<<cdiv((long long)N * nc, 256), 256, 0, stream>>>(
            xh_buf, Kp, off, src, srcLd, srcOff, nc, N);
    };

    auto run_gat = [&](int Kp, const Gat& p, float* outbuf, bool is_coord) {
        const int C = p.fout, M = GAT_HEADS * C, K = p.fin;
        const int Mpad = r64(M);
        // convert + pad weights to f16
        long long wtot = (long long)Mpad * Kp;
        cvt_pad_f16<<<cdiv(wtot, 256), 256, 0, stream>>>(p.W, M, K, wh_buf, Kp, Mpad);
        dim3 gg(cdiv(N, 64), Mpad / 64);
        gemm_f16_wmma<<<gg, 128, 0, stream>>>(xh_buf, Kp, wh_buf, Kp, h_buf, N, M, Kp);

        int t1 = N * GAT_HEADS;
        alpha_kernel<<<cdiv(t1, 256), 256, 0, stream>>>(h_buf, p.as, p.ad, asrc, adst, N, C);
        fill_u32<<<cdiv(t1, 256), 256, 0, stream>>>(emax, 0x007FFFFFu, t1);  // ord(-inf)
        fill_f32<<<cdiv(t1, 256), 256, 0, stream>>>(denom, 0.0f, t1);

        int ET = (E + N) * GAT_HEADS;
        edge_max_kernel<<<cdiv(ET, 256), 256, 0, stream>>>(eidx, E, N, asrc, adst, emax);
        edge_den_kernel<<<cdiv(ET, 256), 256, 0, stream>>>(eidx, E, N, asrc, adst, emax, denom);

        long long accN = (long long)N * M;
        fill_f32<<<cdiv(accN, 256), 256, 0, stream>>>(acc_buf, 0.0f, (int)accN);
        long long tot = (long long)(E + N) * GAT_HEADS * ((C + 3) / 4);
        edge_msg_kernel<<<cdiv(tot, 256), 256, 0, stream>>>(
            eidx, E, N, asrc, adst, emax, denom, h_buf, acc_buf, C);

        if (is_coord)
            finalize_coord<<<cdiv(N, 256), 256, 0, stream>>>(acc_buf, p.b, data, outbuf, N);
        else
            finalize_feat<<<cdiv((long long)N * C, 256), 256, 0, stream>>>(acc_buf, p.b, outbuf, N, C);
    };

    // feat0 = selu(data @ lin_w.T + lin_b)  (N x 256), K=10 -> Kp=64
    {
        begin_x(64);
        put_x(64, 0, data, 10, 0, 10);
        cvt_pad_f16<<<cdiv(256 * 64, 256), 256, 0, stream>>>(lin_w, 256, 10, wh_buf, 64, 256);
        dim3 gg(cdiv(N, 64), 256 / 64);
        gemm_f16_wmma<<<gg, 128, 0, stream>>>(xh_buf, 64, wh_buf, 64, h_buf, N, 256, 64);
        bias_selu_kernel<<<cdiv((long long)N * 256, 256), 256, 0, stream>>>(h_buf, lin_b, feat_buf, N, 256);
    }

    // ---- layer 1: X1 = [coord(2), feat0(256)] -> K=258, Kp=320 ----
    begin_x(320);
    put_x(320, 0, data, 10, 0, 2);
    put_x(320, 2, feat_buf, 256, 0, 256);
    run_gat(320, g[0], coord1, true);
    run_gat(320, g[1], feat_buf, false);     // feat1: 512

    // ---- layer 2: X2 = [coord, coord1, feat1] -> K=516, Kp=576 ----
    begin_x(576);
    put_x(576, 0, data, 10, 0, 2);
    put_x(576, 2, coord1, 2, 0, 2);
    put_x(576, 4, feat_buf, 512, 0, 512);
    run_gat(576, g[2], coord2, true);
    run_gat(576, g[3], feat_buf, false);     // feat2: 256

    // ---- layer 3: X3 = [coord, coord1, coord2, feat2] -> K=262, Kp=320 ----
    begin_x(320);
    put_x(320, 0, data, 10, 0, 2);
    put_x(320, 2, coord1, 2, 0, 2);
    put_x(320, 4, coord2, 2, 0, 2);
    put_x(320, 6, feat_buf, 256, 0, 256);
    run_gat(320, g[4], coord3, true);
    run_gat(320, g[5], feat_buf, false);     // feat3: 128

    // ---- layer 4: X4 = [coord, coord1, coord2, coord3, feat3] -> K=136, Kp=192 ----
    begin_x(192);
    put_x(192, 0, data, 10, 0, 2);
    put_x(192, 2, coord1, 2, 0, 2);
    put_x(192, 4, coord2, 2, 0, 2);
    put_x(192, 6, coord3, 2, 0, 2);
    put_x(192, 8, feat_buf, 128, 0, 128);
    run_gat(192, g[6], (float*)d_out, true); // coord4 -> output
    // d4_feat is dead code (output is coord4 only) — skipped.
    (void)n_in; (void)out_size; (void)ws_size;
}
